// STU_27573690040573
// MI455X (gfx1250) — matrix operand order repaired and checked by reading, not compiled, not run on hardware
//
#include <hip/hip_runtime.h>

// STU spectral filtering as two WMMA GEMM stages (v_wmma_f32_16x16x32_f16 everywhere hot):
//   Stage 1 (proj):  Yt[k][b][d][tau] = (Wk^T @ U[b]^T), Wk = lam_k^0.25 * m_phi block  (f16)
//   Stage 2 (conv):  out[b,t,d] = sum_k sum_{tau<=t} phi_k[t-tau] * Yt[k][b][d][tau]
// Stage 2 operands:
//   A: causal Toeplitz tiles pre-materialized in WMMA A-fragment register order (24x134 tiles, L2-resident)
//   B: 64x256-tau chunks of Yt staged into LDS once per block by the Tensor Data Mover
//      (tensor_load_to_lds + s_wait_tensorcnt, double buffered), fragments read via ds_load_b128.
//      An asm pointer-escape makes the TDM's out-of-band LDS writes visible to the compiler.

typedef __attribute__((ext_vector_type(16))) _Float16     v16h;
typedef __attribute__((ext_vector_type(8)))  _Float16     v8h;
typedef __attribute__((ext_vector_type(8)))  float        v8f;
typedef __attribute__((ext_vector_type(4)))  unsigned int u32x4;
typedef __attribute__((ext_vector_type(8)))  int          i32x8;
typedef __attribute__((ext_vector_type(4)))  int          i32x4;

#define B_     16
#define L_     2048
#define K_     24
#define D_     64
#define NDIAG  134      // delta = rowblk - tau0 in [-96, 2032], multiples of 16
#define CH_    256      // tau per LDS chunk = 8 WMMA k-steps; chunk = 64 x 256 f16 = 32 KB

// ---------------- prep: eig_vecs [L,K] f32 -> phi16 [K][L] f16 ----------------
__global__ void stu_phi_kernel(const float* __restrict__ eig_vecs, _Float16* __restrict__ phi16) {
    int i = blockIdx.x * blockDim.x + threadIdx.x;
    if (i >= K_ * L_) return;
    int k = i / L_, s = i - k * L_;
    phi16[k * L_ + s] = (_Float16)eig_vecs[s * K_ + k];
}

// ---------------- prep: inputs f32 -> Uh f16 ----------------
__global__ void stu_uh_kernel(const float* __restrict__ inp, _Float16* __restrict__ Uh) {
    int i = blockIdx.x * blockDim.x + threadIdx.x;
    if (i >= B_ * L_ * D_) return;
    Uh[i] = (_Float16)inp[i];
}

// ---------------- prep: Wt[k][d][din] = lam_k^0.25 * m_phi[k*64+din][d] (f16) ----------------
__global__ void stu_wt_kernel(const float* __restrict__ m_phi, const float* __restrict__ eig_vals,
                              _Float16* __restrict__ Wt) {
    int i = blockIdx.x * blockDim.x + threadIdx.x;
    if (i >= K_ * D_ * D_) return;
    int din = i & 63;
    int d   = (i >> 6) & 63;
    int k   = i >> 12;
    float s = sqrtf(sqrtf(eig_vals[k]));
    Wt[i] = (_Float16)(s * m_phi[(size_t)((k << 6) + din) * D_ + d]);
}

// ---------------- prep: Toeplitz A-fragments in WMMA register order ----------------
__global__ void stu_atab_kernel(const _Float16* __restrict__ phi16, _Float16* __restrict__ Atab) {
    int idx = blockIdx.x * blockDim.x + threadIdx.x;
    if (idx >= K_ * NDIAG * 32 * 16) return;
    int i    = idx & 15;
    int lane = (idx >> 4) & 31;
    int dk   = idx >> 9;
    int dIdx = dk % NDIAG;
    int k    = dk / NDIAG;
    int delta = dIdx * 16 - 96;
    int l15  = lane & 15;
    int jofs = (lane >> 4) ? 8 : 0;
    int j    = (i < 8) ? (jofs + i) : (jofs + 16 + (i - 8));
    int s    = delta + l15 - j;
    Atab[idx] = (s >= 0 && s < L_) ? phi16[k * L_ + s] : (_Float16)0.f;
}

// ---------------- Stage 1: Yt[k][b][d][tau] via WMMA (Wk^T @ U^T) ----------------
__global__ __launch_bounds__(256) void stu_proj_kernel(const _Float16* __restrict__ Uh,
                                                       const _Float16* __restrict__ Wt,
                                                       _Float16* __restrict__ Yt) {
    const int b    = blockIdx.x;
    const int k    = blockIdx.y;
    const int tid  = threadIdx.x;
    const int wave = tid >> 5;
    const int lane = tid & 31;
    const int half = (lane >> 4) & 1;
    const int l15  = lane & 15;
    const int jofs = half ? 8 : 0;
    const int koff = half ? 16 : 0;

    v16h afr[4][2];
    const _Float16* wbase = Wt + (size_t)k * D_ * D_;
#pragma unroll
    for (int d0 = 0; d0 < 4; ++d0)
#pragma unroll
        for (int ks = 0; ks < 2; ++ks) {
            const _Float16* ap = wbase + (size_t)(d0 * 16 + l15) * D_ + ks * 32 + jofs;
            v8h lo = *(const v8h*)ap;
            v8h hi = *(const v8h*)(ap + 16);
            afr[d0][ks] = __builtin_shufflevector(lo, hi, 0,1,2,3,4,5,6,7,8,9,10,11,12,13,14,15);
        }

    const _Float16* ubase = Uh + (size_t)b * L_ * D_;
    _Float16*       ybase = Yt + (size_t)(k * B_ + b) * D_ * L_;

    for (int tt = wave; tt < L_ / 16; tt += 8) {
        const int tau0 = tt * 16;
        v16h bf[2];
#pragma unroll
        for (int ks = 0; ks < 2; ++ks)
            bf[ks] = __builtin_shufflevector(*(const v8h*)(ubase + (size_t)(tau0 + l15) * D_ + ks * 32 + jofs), *(const v8h*)(ubase + (size_t)(tau0 + l15) * D_ + ks * 32 + jofs + 16), 0,1,2,3,4,5,6,7,8,9,10,11,12,13,14,15);

        v8f c[4];
#pragma unroll
        for (int d0 = 0; d0 < 4; ++d0) c[d0] = (v8f){};
#pragma unroll
        for (int d0 = 0; d0 < 4; ++d0)
#pragma unroll
            for (int ks = 0; ks < 2; ++ks)
                c[d0] = __builtin_amdgcn_wmma_f32_16x16x32_f16(
                    false, afr[d0][ks], false, bf[ks], (short)0, c[d0], false, false);

#pragma unroll
        for (int d0 = 0; d0 < 4; ++d0) {
            const int d = d0 * 16 + half * 8;
#pragma unroll
            for (int r = 0; r < 8; ++r)
                ybase[(size_t)(d + r) * L_ + tau0 + l15] = (_Float16)c[d0][r];
        }
    }
}

// ---------------- TDM: load one 64-row x 512-byte tile (row stride 4096 B) into LDS ----------------
// D# per CDNA5 ISA: group0 = {count|flags, lds_addr, global_addr lo, hi|type=2},
// group1 packs data_size=8B, tensor_dim0=512u, tensor_dim1=64, tile_dim0=64u, tile_dim1=64, stride0=512u.
__device__ __forceinline__ void tdm_load_chunk(const _Float16* gsrc, unsigned lds_byte_off) {
#if __has_builtin(__builtin_amdgcn_tensor_load_to_lds)
    unsigned long long ga = (unsigned long long)(uintptr_t)gsrc;
    u32x4 g0;
    g0[0] = 1u;                                   // count=1, no gather
    g0[1] = lds_byte_off;                         // lds_addr (static LDS base 0: ybuf is sole shared obj)
    g0[2] = (unsigned)(ga & 0xffffffffu);         // global_addr[31:0]
    g0[3] = (unsigned)((ga >> 32) & 0x1ffffffu) | 0x80000000u;  // global_addr[56:32] | type=2
    i32x8 g1;
    g1[0] = (3 << 16);          // workgroup_mask=0 | data_size=3 (8B)
    g1[1] = (512 << 16);        // atomic_barrier_addr=0 | tensor_dim0 lo16 (512 units)
    g1[2] = (64 << 16);         // tensor_dim0 hi16=0 | tensor_dim1 lo16 (64 rows)
    g1[3] = (64 << 16);         // tensor_dim1 hi16=0 | tile_dim0 = 64 units (512 B/row)
    g1[4] = 64;                 // tile_dim1 = 64 rows | tile_dim2 = 0
    g1[5] = 512;                // tensor_dim0_stride lo32 = 512 units (4096 B)
    g1[6] = 0;                  // stride0 hi16 | stride1 lo16
    g1[7] = 0;                  // stride1 hi32
    i32x4 z4 = {0, 0, 0, 0};
#if __clang_major__ >= 23
    i32x8 z8 = {0, 0, 0, 0, 0, 0, 0, 0};
    __builtin_amdgcn_tensor_load_to_lds(g0, g1, z4, z4, z8, 0);
#else
    __builtin_amdgcn_tensor_load_to_lds(g0, g1, z4, z4, 0);
#endif
#endif
}

// ---------------- Stage 2: causal Toeplitz GEMM; B chunks staged via TDM ----------------
__global__ __launch_bounds__(256) void stu_conv_kernel(const _Float16* __restrict__ Yt,
                                                       const _Float16* __restrict__ Atab,
                                                       float* __restrict__ out) {
    __shared__ _Float16 ybuf[2][D_ * CH_];   // 2 x 32 KB double buffer

    const int b    = blockIdx.y;
    const int t0   = blockIdx.x * 128;
    const int tid  = threadIdx.x;
    const int wave = tid >> 5;
    const int lane = tid & 31;
    const int half = (lane >> 4) & 1;
    const int l15  = lane & 15;
    const int koff = half ? 8 : 0;

    const int rowblk = t0 + wave * 16;
    const int nsteps = t0 / 32 + 4;          // causal: skip tau-chunks above the diagonal
    const int nch    = (nsteps + 7) >> 3;    // 256-tau chunks (full chunks always within the 2048 row)

    v8f c[4];
#pragma unroll
    for (int n = 0; n < 4; ++n) c[n] = (v8f){};

    int cur = 0;
    for (int k = 0; k < K_; ++k) {
        const _Float16* ygl   = Yt + (size_t)(k * B_ + b) * D_ * L_;
        const _Float16* abase = Atab + ((size_t)k * NDIAG * 32 + lane) * 16;

#if __has_builtin(__builtin_amdgcn_tensor_load_to_lds)
        if (wave == 0) tdm_load_chunk(ygl, (unsigned)(cur * (D_ * CH_ * 2)));   // prime chunk 0
#endif
        for (int cc = 0; cc < nch; ++cc) {
            __syncthreads();   // everyone done with buf[cur^1] before TDM overwrites it
#if __has_builtin(__builtin_amdgcn_tensor_load_to_lds)
            if (wave == 0) {
                if (cc + 1 < nch) {
                    tdm_load_chunk(ygl + (size_t)(cc + 1) * CH_, (unsigned)((cur ^ 1) * (D_ * CH_ * 2)));
                    __builtin_amdgcn_s_wait_tensorcnt(1);   // chunk cc landed; cc+1 still in flight
                } else {
                    __builtin_amdgcn_s_wait_tensorcnt(0);
                }
            }
#else
            // Fallback: cooperative copy of chunk cc into buf[cur]
            for (int i = tid; i < D_ * (CH_ / 8); i += 256) {
                int r = i / (CH_ / 8), q = i % (CH_ / 8);
                ((v8h*)&ybuf[cur][0])[i] = *(const v8h*)(ygl + (size_t)r * L_ + (size_t)cc * CH_ + q * 8);
            }
#endif
            __syncthreads();   // chunk cc visible in buf[cur]

            // The TDM writes LDS out-of-band (compiler only sees an integer LDS offset).
            // Escape ybuf's address into opaque asm with a memory clobber so the compiler
            // treats the buffer as written and emits real ds_load_b128 reads below.
            {
                _Float16* p = &ybuf[0][0];
                asm volatile("" : "+r"(p)::"memory");
            }

            int scount = nsteps - cc * 8;
            if (scount > 8) scount = 8;
            for (int s = 0; s < scount; ++s) {
                const int tau0 = cc * CH_ + s * 32;
                const int dIdx = (rowblk - tau0 + 96) >> 4;
                const v16h a = *(const v16h*)(abase + (size_t)dIdx * 512);
#pragma unroll
                for (int n = 0; n < 4; ++n) {
                    const v16h bf = __builtin_shufflevector(*(const v8h*)&ybuf[cur][(n * 16 + l15) * CH_ + s * 32 + koff], *(const v8h*)&ybuf[cur][(n * 16 + l15) * CH_ + s * 32 + koff + 16], 0,1,2,3,4,5,6,7,8,9,10,11,12,13,14,15);
                    c[n] = __builtin_amdgcn_wmma_f32_16x16x32_f16(
                        false, a, false, bf, (short)0, c[n], false, false);
                }
            }
            cur ^= 1;
        }
    }

    // Epilogue: f32 C layout -> out[b][t][d]
    const int rbase = rowblk + (half ? 8 : 0);
#pragma unroll
    for (int n = 0; n < 4; ++n) {
        const int col = n * 16 + l15;
#pragma unroll
        for (int r = 0; r < 8; ++r)
            out[((size_t)b * L_ + rbase + r) * D_ + col] = c[n][r];
    }
}

extern "C" void kernel_launch(void* const* d_in, const int* in_sizes, int n_in,
                              void* d_out, int out_size, void* d_ws, size_t ws_size,
                              hipStream_t stream) {
    const float* inputs   = (const float*)d_in[0];  // [B, L, D]
    const float* m_phi    = (const float*)d_in[1];  // [K*D, D]
    const float* eig_vals = (const float*)d_in[2];  // [K]
    const float* eig_vecs = (const float*)d_in[3];  // [L, K]
    float* out = (float*)d_out;                     // [B, L, D]

    char* ws = (char*)d_ws;
    _Float16* Yt    = (_Float16*)ws;                                            // 100,663,296 B
    _Float16* phi16 = (_Float16*)(ws + (size_t)100663296);                      //      98,304 B
    _Float16* Atab  = (_Float16*)(ws + (size_t)100663296 + 98304);              //   3,293,184 B
    _Float16* Uh    = (_Float16*)(ws + (size_t)100663296 + 98304 + 3293184);    //   4,194,304 B
    _Float16* Wt    = (_Float16*)(ws + (size_t)100663296 + 98304 + 3293184 + 4194304); // 196,608 B

    stu_phi_kernel<<<(K_ * L_ + 255) / 256, 256, 0, stream>>>(eig_vecs, phi16);
    stu_uh_kernel<<<(B_ * L_ * D_ + 255) / 256, 256, 0, stream>>>(inputs, Uh);
    stu_wt_kernel<<<(K_ * D_ * D_ + 255) / 256, 256, 0, stream>>>(m_phi, eig_vals, Wt);
    stu_atab_kernel<<<(K_ * NDIAG * 32 * 16 + 255) / 256, 256, 0, stream>>>(phi16, Atab);
    stu_proj_kernel<<<dim3(B_, K_), 256, 0, stream>>>(Uh, Wt, Yt);
    stu_conv_kernel<<<dim3(L_ / 128, B_), 256, 0, stream>>>(Yt, Atab, out);
}
